// BitNetAttention_23106924052908
// MI455X (gfx1250) — compile-verified, matrix operand order checked by reading, not run on hardware
//
#include <hip/hip_runtime.h>
#include <hip/hip_bf16.h>
#include <cstdint>
#include <cstddef>

// ---------------------------------------------------------------------------
// Types for WMMA fragments (gfx1250 wave32)
// ---------------------------------------------------------------------------
typedef __bf16 v16bf __attribute__((ext_vector_type(16)));
typedef float  v8f   __attribute__((ext_vector_type(8)));
typedef float  f4v   __attribute__((ext_vector_type(4)));
typedef unsigned short u16v8 __attribute__((ext_vector_type(8)));

struct P32 { f4v a, b; };   // 32 bytes == sizeof(v16bf)

// Problem constants
constexpr int B_   = 2;
constexpr int S_   = 2048;
constexpr int H_   = 768;
constexpr int NH   = 12;
constexpr int NKV  = 6;
constexpr int HD   = 64;
constexpr int M_   = B_ * S_;             // 4096 activation rows
constexpr int NQKV = H_ + 2 * (NKV * HD); // 1536
constexpr float ALPHA = 0.7f;

// ---------------------------------------------------------------------------
// Helpers
// ---------------------------------------------------------------------------
// 16-bit A/B fragment load (row-major source): lane holds row (lane&15) of the
// tile; K-chunks at k + 8*half and k + 16 + 8*half (half = lane>>4).
// `rowp` must point at element [tile_row + (lane&15)][k0].
__device__ __forceinline__ v16bf load_frag(const unsigned short* rowp, int lh) {
  P32 t;
  t.a = *reinterpret_cast<const f4v*>(rowp + 8 * lh);
  t.b = *reinterpret_cast<const f4v*>(rowp + 16 + 8 * lh);
  return __builtin_bit_cast(v16bf, t);
}

__device__ __forceinline__ unsigned short f2bf(float f) {
  unsigned u = __float_as_uint(f);
  u += 0x7fffu + ((u >> 16) & 1u);     // round-to-nearest-even
  return (unsigned short)(u >> 16);
}

// ---------------------------------------------------------------------------
// 16(M) x 64(N) WMMA tile over K.  Deliberately NOT software-pipelined with
// explicit double buffers: that version kept ~10 fragments (160 VGPRs) live
// and spilled to scratch in the mainloop.  This shape allocates cleanly; the
// scheduler + s_clause overlap the four B-fragment loads with the WMMAs.
// ---------------------------------------------------------------------------
__device__ __forceinline__ void gemm_tile(const unsigned short* __restrict__ xrow,
                                          const unsigned short* __restrict__ W,
                                          int K, int n0, int rr, int lh,
                                          v8f acc[4]) {
  for (int k0 = 0; k0 < K; k0 += 32) {
    __builtin_prefetch(xrow + k0 + 64, 0, 1);
    v16bf a = load_frag(xrow + k0, lh);
#pragma unroll
    for (int j = 0; j < 4; ++j) {
      const unsigned short* wrow = W + (size_t)(n0 + j * 16 + rr) * K + k0;
      v16bf b = load_frag(wrow, lh);
      acc[j] = __builtin_amdgcn_wmma_f32_16x16x32_bf16(
          false, a, false, b, (short)0, acc[j], false, false);
    }
  }
}

// ---------------------------------------------------------------------------
// 1) sum(|w|) — single block, deterministic
// ---------------------------------------------------------------------------
__global__ void absum_kernel(const float* __restrict__ w, int n, float* __restrict__ out) {
  __shared__ float sm[256];
  float s = 0.f;
  for (int i = threadIdx.x; i < n; i += 256) s += fabsf(w[i]);
  sm[threadIdx.x] = s;
  __syncthreads();
  for (int d = 128; d > 0; d >>= 1) {
    if ((int)threadIdx.x < d) sm[threadIdx.x] += sm[threadIdx.x + d];
    __syncthreads();
  }
  if (threadIdx.x == 0) out[0] = sm[0];
}

// ---------------------------------------------------------------------------
// 2) ternarize weights -> bf16 ; cast activations -> bf16
// ---------------------------------------------------------------------------
__global__ void quant_kernel(const float* __restrict__ w, int n,
                             const float* __restrict__ sum, float inv_n,
                             unsigned short* __restrict__ out) {
  int i = blockIdx.x * 256 + threadIdx.x;
  if (i >= n) return;
  float thr = ALPHA * sum[0] * inv_n;
  float x = w[i];
  float q = (fabsf(x) >= thr) ? ((x > 0.f) ? 1.f : -1.f) : 0.f;
  out[i] = f2bf(q);
}

__global__ void cast_bf16_kernel(const float* __restrict__ x, int n,
                                 unsigned short* __restrict__ out) {
  int i = blockIdx.x * 256 + threadIdx.x;
  if (i >= n) return;
  out[i] = f2bf(x[i]);
}

// ---------------------------------------------------------------------------
// 3) Fused QKV projection + RoPE + scale + layout epilogue.
//    Y = Xb @ Wcat^T ; each wave's 64-col tile is exactly one head
//    (head = n0/64).  RoPE pair (d, d+32) == acc tiles (j, j+2), same lane,
//    same element => fully in-register.  Writes:
//      Q -> Qb[b,NH,S,HD]  (scale = mean|Wq| * rsqrt(64) folded; rotation
//                           commutes with the scalar)
//      K -> Kb[b,NKV,S,HD] (scale = mean|Wk|)
//      V -> Vt[b,NKV,HD,S] (scale = mean|Wv|, transposed for PV A-fragments)
// ---------------------------------------------------------------------------
__global__ void gemm_qkv_rope_kernel(const unsigned short* __restrict__ X,
                                     const unsigned short* __restrict__ W,
                                     unsigned short* __restrict__ Qb,
                                     unsigned short* __restrict__ Kb,
                                     unsigned short* __restrict__ Vt,
                                     const float* __restrict__ sums) {
  const int lane = threadIdx.x & 31;
  const int wid  = threadIdx.x >> 5;
  const int m0 = blockIdx.y * 128 + wid * 16;
  const int n0 = blockIdx.x * 64;
  const int rr = lane & 15;
  const int lh = lane >> 4;

  v8f acc[4] = {};
  gemm_tile(X + (size_t)(m0 + rr) * H_, W, H_, n0, rr, lh, acc);

  const int head = n0 >> 6;                 // 0..23 : Q 0-11, K 12-17, V 18-23
  if (head < NH + NKV) {
    // --- Q or K: RoPE + scale ---
    const bool isQ = head < NH;
    const int hh = isQ ? head : (head - NH);
    const int heads = isQ ? NH : NKV;
    unsigned short* dst = isQ ? Qb : Kb;
    const float sc = isQ ? (sums[0] * (1.0f / 589824.0f) * 0.125f)
                         : (sums[1] * (1.0f / 294912.0f));
#pragma unroll
    for (int j = 0; j < 2; ++j) {
      const int i = j * 16 + rr;                       // 0..31
      const float fr = __powf(10000.0f, -(float)i / 32.0f);
#pragma unroll
      for (int t = 0; t < 8; ++t) {
        const int m = m0 + t + 8 * lh;
        const int b = m >> 11, pos = m & (S_ - 1);
        const float ang = (float)pos * fr;
        const float c = __cosf(ang), s = __sinf(ang);
        const float q0 = acc[j][t] * sc, q1 = acc[j + 2][t] * sc;
        const size_t o = ((size_t)(b * heads + hh) * S_ + pos) * HD + i;
        dst[o]      = f2bf(q0 * c - q1 * s);
        dst[o + 32] = f2bf(q1 * c + q0 * s);
      }
    }
  } else {
    // --- V: scale + transpose ---
    const int kv = head - NH - NKV;
    const float sc = sums[2] * (1.0f / 294912.0f);
#pragma unroll
    for (int j = 0; j < 4; ++j) {
      const int d = j * 16 + rr;
#pragma unroll
      for (int t = 0; t < 8; ++t) {
        const int m = m0 + t + 8 * lh;
        const int b = m >> 11, pos = m & (S_ - 1);
        Vt[((size_t)(b * NKV + kv) * HD + d) * S_ + pos] = f2bf(acc[j][t] * sc);
      }
    }
  }
}

// ---------------------------------------------------------------------------
// 4) Output projection: f32 out = Ob @ Woq^T * mean|Wo|
// ---------------------------------------------------------------------------
__global__ void gemm_out_kernel(const unsigned short* __restrict__ X,
                                const unsigned short* __restrict__ W,
                                float* __restrict__ Y, int K, int N,
                                const float* __restrict__ sum_ptr, float inv_nw) {
  const int lane = threadIdx.x & 31;
  const int wid  = threadIdx.x >> 5;
  const int m0 = blockIdx.y * 128 + wid * 16;
  const int n0 = blockIdx.x * 64;
  const int rr = lane & 15;
  const int lh = lane >> 4;

  v8f acc[4] = {};
  gemm_tile(X + (size_t)(m0 + rr) * K, W, K, n0, rr, lh, acc);

  const float sc = sum_ptr[0] * inv_nw;
#pragma unroll
  for (int j = 0; j < 4; ++j)
#pragma unroll
    for (int t = 0; t < 8; ++t)
      Y[(size_t)(m0 + t + 8 * lh) * N + (n0 + j * 16 + rr)] = acc[j][t] * sc;
}

// ---------------------------------------------------------------------------
// 5) Flash-style attention (no mask).  One wave per 16-query tile.
//    S^T = K_tile @ Q^T (softmax axis -> 8 in-lane regs + one shfl_xor 16),
//    C-layout of S^T == B-fragment layout of P (no cross-lane movement),
//    O^T += V^T_tile @ P.  GQA: kv head = h/2.  V^T fragment loads are issued
//    right after the QK^T WMMAs (K fragments dead by then) so they overlap the
//    softmax VALU work without inflating register lifetimes.
// ---------------------------------------------------------------------------
__global__ void attn_kernel(const unsigned short* __restrict__ Qb,
                            const unsigned short* __restrict__ Kb,
                            const unsigned short* __restrict__ Vt,
                            unsigned short* __restrict__ O) {
  const int lane = threadIdx.x & 31;
  const int wid  = threadIdx.x >> 5;
  const int s0 = (blockIdx.x * 4 + wid) * 16;
  const int h = blockIdx.y, b = blockIdx.z;
  const int kv = h >> 1;
  const int rr = lane & 15;
  const int lh = lane >> 4;

  const unsigned short* Qh = Qb + (size_t)(b * NH + h) * S_ * HD;
  const unsigned short* Kh = Kb + (size_t)(b * NKV + kv) * S_ * HD;
  const unsigned short* Vh = Vt + (size_t)(b * NKV + kv) * HD * S_;

  // Q as B-fragment (Q^T is the 64x16 B matrix): lane rr = query column s.
  v16bf qf0 = load_frag(Qh + (size_t)(s0 + rr) * HD + 0,  lh);
  v16bf qf1 = load_frag(Qh + (size_t)(s0 + rr) * HD + 32, lh);

  v8f o[4] = {};
  float mrun = -1e30f, lrun = 0.f;

  for (int t0 = 0; t0 < S_; t0 += 32) {
    const unsigned short* krow0 = Kh + (size_t)(t0 + rr) * HD;
    const unsigned short* krow1 = Kh + (size_t)(t0 + 16 + rr) * HD;
    v16bf k00 = load_frag(krow0 + 0,  lh);
    v16bf k01 = load_frag(krow0 + 32, lh);
    v16bf k10 = load_frag(krow1 + 0,  lh);
    v16bf k11 = load_frag(krow1 + 32, lh);
    __builtin_prefetch(krow0 + 32 * HD, 0, 1);  // next step's K rows

    v8f st0 = {}, st1 = {};
    st0 = __builtin_amdgcn_wmma_f32_16x16x32_bf16(false, k00, false, qf0, (short)0, st0, false, false);
    st0 = __builtin_amdgcn_wmma_f32_16x16x32_bf16(false, k01, false, qf1, (short)0, st0, false, false);
    st1 = __builtin_amdgcn_wmma_f32_16x16x32_bf16(false, k10, false, qf0, (short)0, st1, false, false);
    st1 = __builtin_amdgcn_wmma_f32_16x16x32_bf16(false, k11, false, qf1, (short)0, st1, false, false);
    // st* C-layout: lane -> query s = lane&15 ; element t -> key (t + 8*lh [+16])

    // V^T fragments for this step: in flight during the softmax VALU below.
    v16bf vf[4];
#pragma unroll
    for (int dt = 0; dt < 4; ++dt)
      vf[dt] = load_frag(Vh + (size_t)(dt * 16 + rr) * S_ + t0, lh);

    float mx = -1e30f;
#pragma unroll
    for (int t = 0; t < 8; ++t) { mx = fmaxf(mx, st0[t]); mx = fmaxf(mx, st1[t]); }
    mx = fmaxf(mx, __shfl_xor(mx, 16, 32));
    float mnew = fmaxf(mrun, mx);

    float p0[8], p1[8], rs = 0.f;
#pragma unroll
    for (int t = 0; t < 8; ++t) {
      p0[t] = __expf(st0[t] - mnew);
      p1[t] = __expf(st1[t] - mnew);
      rs += p0[t] + p1[t];
    }
    rs += __shfl_xor(rs, 16, 32);

    float alpha = __expf(mrun - mnew);
    lrun = lrun * alpha + rs;
    mrun = mnew;
#pragma unroll
    for (int j = 0; j < 4; ++j)
#pragma unroll
      for (int t = 0; t < 8; ++t) o[j][t] *= alpha;

    // Pack P into B-fragment: elem j = key 8*lh+j (st0[j], this lane),
    // elem 8+j = key 16+8*lh+j (st1[j], this lane).  No cross-lane movement.
    v16bf pf;
#pragma unroll
    for (int t = 0; t < 8; ++t) {
      pf[t]     = (__bf16)p0[t];
      pf[t + 8] = (__bf16)p1[t];
    }

#pragma unroll
    for (int dt = 0; dt < 4; ++dt)
      o[dt] = __builtin_amdgcn_wmma_f32_16x16x32_bf16(false, vf[dt], false, pf, (short)0, o[dt], false, false);
  }

  float inv = 1.0f / lrun;
  // O^T C-layout: lane -> s = lane&15 ; element t -> d = dt*16 + t + 8*lh
#pragma unroll
  for (int dt = 0; dt < 4; ++dt) {
    u16v8 wv;
#pragma unroll
    for (int t = 0; t < 8; ++t) wv[t] = f2bf(o[dt][t] * inv);
    size_t oidx = ((size_t)(b * S_ + s0 + rr) * NH + h) * HD + dt * 16 + 8 * lh;
    *reinterpret_cast<u16v8*>(O + oidx) = wv;
  }
}

// ---------------------------------------------------------------------------
// Host-side launch
// ---------------------------------------------------------------------------
extern "C" void kernel_launch(void* const* d_in, const int* in_sizes, int n_in,
                              void* d_out, int out_size, void* d_ws, size_t ws_size,
                              hipStream_t stream) {
  const float* x  = (const float*)d_in[0];   // [2,2048,768]
  const float* Wq = (const float*)d_in[1];   // [768,768]
  const float* Wk = (const float*)d_in[2];   // [384,768]
  const float* Wv = (const float*)d_in[3];   // [384,768]
  const float* Wo = (const float*)d_in[4];   // [768,768]
  float* out = (float*)d_out;                // [2,2048,768] f32

  char* ws = (char*)d_ws;
  // workspace layout (256B-aligned offsets) — ~27.4 MB total
  float*          sums = (float*)(ws + 0);                   // 4 f32
  unsigned short* Wcat = (unsigned short*)(ws + 256);        // [1536,768] bf16
  unsigned short* Woq  = (unsigned short*)(ws + 2359552);    // [768,768]  bf16
  unsigned short* Xb   = (unsigned short*)(ws + 3539200);    // [4096,768] bf16
  unsigned short* Qb   = (unsigned short*)(ws + 9830656);    // [2,12,2048,64] bf16
  unsigned short* Kb   = (unsigned short*)(ws + 16122112);   // [2,6,2048,64] bf16
  unsigned short* Vt   = (unsigned short*)(ws + 19267840);   // [2,6,64,2048] bf16
  unsigned short* Ob   = (unsigned short*)(ws + 22413568);   // [4096,768] bf16

  const int nWq = 768 * 768, nWk = 384 * 768, nWv = 384 * 768, nWo = 768 * 768;
  const int nX  = M_ * H_;

  // 1) weight abs-sums (deterministic single-block reductions)
  absum_kernel<<<1, 256, 0, stream>>>(Wq, nWq, sums + 0);
  absum_kernel<<<1, 256, 0, stream>>>(Wk, nWk, sums + 1);
  absum_kernel<<<1, 256, 0, stream>>>(Wv, nWv, sums + 2);
  absum_kernel<<<1, 256, 0, stream>>>(Wo, nWo, sums + 3);

  // 2) ternarize weights into concatenated [Wq;Wk;Wv] and Wo ; cast X
  quant_kernel<<<(nWq + 255) / 256, 256, 0, stream>>>(Wq, nWq, sums + 0, 1.0f / nWq, Wcat);
  quant_kernel<<<(nWk + 255) / 256, 256, 0, stream>>>(Wk, nWk, sums + 1, 1.0f / nWk, Wcat + nWq);
  quant_kernel<<<(nWv + 255) / 256, 256, 0, stream>>>(Wv, nWv, sums + 2, 1.0f / nWv, Wcat + nWq + nWk);
  quant_kernel<<<(nWo + 255) / 256, 256, 0, stream>>>(Wo, nWo, sums + 3, 1.0f / nWo, Woq);
  cast_bf16_kernel<<<(nX + 255) / 256, 256, 0, stream>>>(x, nX, Xb);

  // 3) fused QKV projection + RoPE + scale + layout (no f32 intermediate)
  {
    dim3 grid(NQKV / 64, M_ / 128);
    gemm_qkv_rope_kernel<<<grid, 256, 0, stream>>>(Xb, Wcat, Qb, Kb, Vt, sums);
  }

  // 4) attention
  {
    dim3 grid(S_ / 64, NH, B_);
    attn_kernel<<<grid, 128, 0, stream>>>(Qb, Kb, Vt, Ob);
  }

  // 5) output projection with mean|Wo| epilogue -> f32 d_out
  {
    dim3 grid(H_ / 64, M_ / 128);
    gemm_out_kernel<<<grid, 256, 0, stream>>>(Ob, Woq, out, H_, H_, sums + 3, 1.0f / nWo);
  }
}
